// GRetrieverModel_49108656063009
// MI455X (gfx1250) — compile-verified, matrix operand order checked by reading, not run on hardware
//
#include <hip/hip_runtime.h>
#include <hip/hip_bf16.h>

// ---------------- CDNA5 WMMA types ----------------
typedef __bf16  v16bf __attribute__((ext_vector_type(16)));
typedef float   v8f   __attribute__((ext_vector_type(8)));

#define WMMA_BF16(a, b, c) \
    __builtin_amdgcn_wmma_f32_16x16x32_bf16(false, (a), false, (b), (short)0, (c), false, false)

// Async global->LDS path (ASYNCcnt), gated on toolchain support.
#if defined(__has_builtin)
#if __has_builtin(__builtin_amdgcn_global_load_async_to_lds_b128) && \
    __has_builtin(__builtin_amdgcn_s_wait_asynccnt)
#define USE_ASYNC_LDS 1
#endif
#endif

// b128 payload type as expected by the async builtin (from hipcc diagnostic):
// param 0: v4i __attribute__((address_space(1)))*   (global, prints as __device__)
// param 1: v4i __attribute__((address_space(3)))*   (LDS, prints as __shared__)
typedef int v4i __attribute__((vector_size(4 * sizeof(int))));
typedef v4i __attribute__((address_space(1)))* g_b128_ptr;
typedef v4i __attribute__((address_space(3)))* l_b128_ptr;

static constexpr int NN   = 100000;   // nodes
static constexpr int EE   = 1600000;  // edges (before self loops)
static constexpr int DD   = 768;      // node embed dim
static constexpr int HC   = 256;      // H * C
static constexpr int HIDD = 128;

// ---------------- helpers ----------------
__device__ __forceinline__ unsigned ordf(float f) {
    unsigned u = __float_as_uint(f);
    return (u & 0x80000000u) ? ~u : (u | 0x80000000u);
}
__device__ __forceinline__ float deordf(unsigned e) {
    unsigned u = (e & 0x80000000u) ? (e ^ 0x80000000u) : ~e;
    return __uint_as_float(u);
}

// ---------------- A-panel staging (16 rows x K) into LDS as bf16 ----------------
// f32 source: convert on the fly through VGPRs.
__device__ __forceinline__ void stage_panel(const float* __restrict__ A, __bf16* sA,
                                            int rowBase, int K, int M) {
    for (int idx = threadIdx.x; idx < 16 * K; idx += 256) {
        int r = idx / K, c = idx - r * K;
        int gr = rowBase + r;
        float v = (gr < M) ? A[(size_t)gr * K + c] : 0.0f;
        sA[r * K + c] = (__bf16)v;
    }
}
// bf16 source: panel is one contiguous byte range (M % 16 == 0) -> async DMA to LDS.
__device__ __forceinline__ void stage_panel(const __bf16* __restrict__ A, __bf16* sA,
                                            int rowBase, int K, int M) {
    (void)M;
    const int totalBytes = 16 * K * 2;
#if USE_ASYNC_LDS
    const char* g = (const char*)(A + (size_t)rowBase * K);
    char* l = (char*)sA;
    for (int b = threadIdx.x * 16; b < totalBytes; b += 256 * 16) {
        __builtin_amdgcn_global_load_async_to_lds_b128(
            (g_b128_ptr)(g + b), (l_b128_ptr)(l + b), 0, 0);
    }
    __builtin_amdgcn_s_wait_asynccnt(0);
#else
    const int totalElems = totalBytes / 2;
    for (int idx = threadIdx.x * 8; idx < totalElems; idx += 256 * 8) {
        // 8 bf16 = 16B chunks; K % 8 == 0 so chunks never cross a row
        *(int4*)&sA[idx] = *(const int4*)&A[(size_t)rowBase * K + idx];
    }
#endif
}

// ---------------- weight transpose + bf16 convert: W[K,Nc] -> Wt[Nc,K] ----------------
__global__ void k_wt(const float* __restrict__ W, __bf16* __restrict__ Wt, int K, int Nc) {
    int idx = blockIdx.x * 256 + threadIdx.x;
    if (idx < K * Nc) {
        int k = idx / Nc, n = idx % Nc;
        Wt[(size_t)n * K + k] = (__bf16)W[idx];
    }
}

// ---------------- WMMA GEMM: C[M,Nc] = A[M,K] x Wt[Nc,K]^T ----------------
// Block = 256 threads = 8 waves; block computes 16 rows x Nc cols.
template <typename TA>
__global__ __launch_bounds__(256) void k_gemm(const TA* __restrict__ A,
                                              const __bf16* __restrict__ Wt,
                                              float* __restrict__ C,
                                              int M, int K, int Nc) {
    __shared__ __bf16 sA[16 * 768];
    const int rowBase = blockIdx.x * 16;
    const int wave = threadIdx.x >> 5, lane = threadIdx.x & 31;
    const int tpw = Nc >> 7;  // col-tiles per wave: Nc=256 -> 2, Nc=128 -> 1

    // Prefetch this wave's B panel (L2 resident, gfx1250 global_prefetch)
    for (int t = 0; t < tpw; ++t) {
        int col = (wave * tpw + t) * 16 + (lane & 15);
        __builtin_prefetch(&Wt[(size_t)col * K], 0, 3);
    }

    stage_panel(A, sA, rowBase, K, M);
    __syncthreads();

    const int rA = lane & 15;
    const int kHi = (lane >> 4) << 4;
    for (int t = 0; t < tpw; ++t) {
        const int colBase = (wave * tpw + t) * 16;
        const __bf16* bRow = &Wt[(size_t)(colBase + (lane & 15)) * K];
        v8f acc = {};
        for (int k = 0; k < K; k += 32) {
            v16bf a = *(const v16bf*)&sA[rA * K + k + kHi];
            v16bf b = *(const v16bf*)&bRow[k + kHi];
            acc = WMMA_BF16(a, b, acc);
        }
        // C/D layout: VGPR j, lanes 0-15 -> M=j, lanes 16-31 -> M=j+8; N = lane&15
        const int m0 = rowBase + ((lane >> 4) << 3);
        const int n  = colBase + (lane & 15);
        #pragma unroll
        for (int j = 0; j < 8; ++j) {
            int m = m0 + j;
            if (m < M) C[(size_t)m * Nc + n] = acc[j];
        }
    }
}

// ---------------- attention coefficients: wave per node ----------------
__global__ __launch_bounds__(256) void k_alpha(const float* __restrict__ H,
                                               const float* __restrict__ a_src,
                                               const float* __restrict__ a_dst,
                                               float* __restrict__ AS, float* __restrict__ AD,
                                               int Nn) {
    int node = (blockIdx.x * 256 + threadIdx.x) >> 5;
    int lane = threadIdx.x & 31;
    if (node >= Nn) return;
    const float* hp = H + (size_t)node * HC + lane * 8;
    float s = 0.f, d = 0.f;
    #pragma unroll
    for (int i = 0; i < 8; ++i) {
        float hv = hp[i];
        s += hv * a_src[lane * 8 + i];
        d += hv * a_dst[lane * 8 + i];
    }
    #pragma unroll
    for (int off = 1; off < 16; off <<= 1) {
        s += __shfl_xor(s, off, 32);
        d += __shfl_xor(d, off, 32);
    }
    int head = lane >> 4;
    if ((lane & 15) == 0) {
        AS[node * 2 + head] = s;
        AD[node * 2 + head] = d;
    }
}

// ---------------- init accumulators ----------------
__global__ void k_init(float* __restrict__ HACC, unsigned* __restrict__ SEGMAX,
                       float* __restrict__ SEGSUM, int Nn) {
    int i = blockIdx.x * 256 + threadIdx.x;
    if (i < Nn * HC) HACC[i] = 0.0f;
    if (i < Nn * 2) { SEGMAX[i] = 0u; SEGSUM[i] = 0.0f; }  // 0u == ordered(-inf)
}

// ---------------- edge pass 1: leaky-relu logits + segment max ----------------
__global__ void k_elog(const int* __restrict__ ei, const float* __restrict__ AS,
                       const float* __restrict__ AD, float* __restrict__ ELOG,
                       unsigned* __restrict__ SEGMAX, int Eo, int Nn) {
    int e = blockIdx.x * 256 + threadIdx.x;
    if (e >= Eo + Nn) return;
    int s, d;
    if (e < Eo) { s = ei[e]; d = ei[Eo + e]; } else { s = d = e - Eo; }
    #pragma unroll
    for (int h = 0; h < 2; ++h) {
        float v = AS[s * 2 + h] + AD[d * 2 + h];
        v = (v > 0.f) ? v : 0.2f * v;
        ELOG[(size_t)e * 2 + h] = v;
        atomicMax(&SEGMAX[d * 2 + h], ordf(v));
    }
}

// ---------------- edge pass 2: exp + segment sum ----------------
__global__ void k_expsum(const int* __restrict__ ei, float* __restrict__ ELOG,
                         const unsigned* __restrict__ SEGMAX, float* __restrict__ SEGSUM,
                         int Eo, int Nn) {
    int e = blockIdx.x * 256 + threadIdx.x;
    if (e >= Eo + Nn) return;
    int d = (e < Eo) ? ei[Eo + e] : (e - Eo);
    #pragma unroll
    for (int h = 0; h < 2; ++h) {
        float v = ELOG[(size_t)e * 2 + h];
        float m = deordf(SEGMAX[d * 2 + h]);
        float ex = __expf(v - m);
        ELOG[(size_t)e * 2 + h] = ex;
        unsafeAtomicAdd(&SEGSUM[d * 2 + h], ex);
    }
}

// ---------------- edge pass 3: weighted scatter-add (wave per edge) ----------------
__global__ __launch_bounds__(256) void k_agg(const int* __restrict__ ei,
                                             const float* __restrict__ HLIN,
                                             const float* __restrict__ ELOG,
                                             const float* __restrict__ SEGSUM,
                                             float* __restrict__ HACC, int Eo, int Nn) {
    int e = (blockIdx.x * 256 + threadIdx.x) >> 5;
    int lane = threadIdx.x & 31;
    if (e >= Eo + Nn) return;
    int s, d;
    if (e < Eo) { s = ei[e]; d = ei[Eo + e]; } else { s = d = e - Eo; }
    int head = lane >> 4;
    float alpha = ELOG[(size_t)e * 2 + head] / (SEGSUM[d * 2 + head] + 1e-16f);
    const float* hp = HLIN + (size_t)s * HC + lane * 8;
    float* op = HACC + (size_t)d * HC + lane * 8;
    #pragma unroll
    for (int i = 0; i < 8; ++i) unsafeAtomicAdd(&op[i], hp[i] * alpha);
}

// ---------------- bias + relu + bf16 convert ----------------
__global__ void k_brelu(const float* __restrict__ HACC, const float* __restrict__ b,
                        __bf16* __restrict__ HB, int Nn) {
    int i = blockIdx.x * 256 + threadIdx.x;
    if (i >= Nn * HC) return;
    float v = HACC[i] + b[i & (HC - 1)];
    HB[i] = (__bf16)(v > 0.f ? v : 0.f);
}

// ---------------- question projection (tiny) ----------------
__global__ void k_qemb(const float* __restrict__ q, const float* __restrict__ Wq,
                       const float* __restrict__ bq, float* __restrict__ QEMB) {
    int j = threadIdx.x;  // 128 threads
    float acc = bq[j];
    for (int i = 0; i < DD; ++i) acc += q[i] * Wq[i * HIDD + j];
    QEMB[j] = acc;
}
__global__ void k_qvec(const float* __restrict__ QEMB, const float* __restrict__ Ws1,
                       const float* __restrict__ bs1, float* __restrict__ QVEC) {
    int j = threadIdx.x;  // 128 threads
    float acc = bs1[j];
    for (int i = 0; i < HIDD; ++i) acc += QEMB[i] * Ws1[(HIDD + i) * HIDD + j];
    QVEC[j] = acc;
}

// ---------------- fused scorer: h2 @ Wn + bn -> @ Ws1 + qvec -> relu -> dot ws2 ----------------
__global__ __launch_bounds__(256) void k_final(const __bf16* __restrict__ HB,
                                               const __bf16* __restrict__ Wnt,
                                               const float* __restrict__ bn,
                                               const __bf16* __restrict__ Ws1t,
                                               const float* __restrict__ qvec,
                                               const float* __restrict__ ws2,
                                               const float* __restrict__ bs2,
                                               float* __restrict__ out, int Nn) {
    __shared__ __bf16 sM[8][16 * 128];
    __shared__ __bf16 sH[8][16 * 128];
    const int wave = threadIdx.x >> 5, lane = threadIdx.x & 31;
    const int nodeBase = (blockIdx.x * 8 + wave) * 16;
    const bool valid = (nodeBase + 16) <= Nn;  // N % 16 == 0
    const int kHi = (lane >> 4) << 4;
    const int rowA = valid ? (nodeBase + (lane & 15)) : 0;

    // stage 1: node_emb = h2 @ Wnt + bn  (16x128, K=256)
    for (int ct = 0; ct < 8; ++ct) {
        v8f acc = {};
        const __bf16* bRow = &Wnt[(size_t)(ct * 16 + (lane & 15)) * HC];
        for (int k = 0; k < HC; k += 32) {
            v16bf a = *(const v16bf*)&HB[(size_t)rowA * HC + k + kHi];
            v16bf b = *(const v16bf*)&bRow[k + kHi];
            acc = WMMA_BF16(a, b, acc);
        }
        #pragma unroll
        for (int j = 0; j < 8; ++j) {
            int m = j + ((lane >> 4) << 3);
            int n = ct * 16 + (lane & 15);
            sM[wave][m * 128 + n] = (__bf16)(acc[j] + bn[n]);
        }
    }
    __syncthreads();

    // stage 2: hmid = relu(node_emb @ Ws1t + qvec)  (16x128, K=128)
    for (int ct = 0; ct < 8; ++ct) {
        v8f acc = {};
        const __bf16* bRow = &Ws1t[(size_t)(ct * 16 + (lane & 15)) * HIDD];
        for (int k = 0; k < HIDD; k += 32) {
            v16bf a = *(const v16bf*)&sM[wave][(lane & 15) * 128 + k + kHi];
            v16bf b = *(const v16bf*)&bRow[k + kHi];
            acc = WMMA_BF16(a, b, acc);
        }
        #pragma unroll
        for (int j = 0; j < 8; ++j) {
            int m = j + ((lane >> 4) << 3);
            int n = ct * 16 + (lane & 15);
            float v = acc[j] + qvec[n];
            sH[wave][m * 128 + n] = (__bf16)(v > 0.f ? v : 0.f);
        }
    }
    __syncthreads();

    // stage 3: scores
    if (valid && lane < 16) {
        float sc = bs2[0];
        for (int n = 0; n < 128; ++n) sc += (float)sH[wave][lane * 128 + n] * ws2[n];
        out[nodeBase + lane] = sc;
    }
}

// ==================================================================
extern "C" void kernel_launch(void* const* d_in, const int* in_sizes, int n_in,
                              void* d_out, int out_size, void* d_ws, size_t ws_size,
                              hipStream_t stream) {
    const float* x        = (const float*)d_in[0];
    const int*   ei       = (const int*)d_in[1];
    const float* q        = (const float*)d_in[2];
    const float* W1       = (const float*)d_in[3];
    const float* a_src1   = (const float*)d_in[4];
    const float* a_dst1   = (const float*)d_in[5];
    const float* b1       = (const float*)d_in[6];
    const float* W2       = (const float*)d_in[7];
    const float* a_src2   = (const float*)d_in[8];
    const float* a_dst2   = (const float*)d_in[9];
    const float* b2       = (const float*)d_in[10];
    const float* Wn       = (const float*)d_in[11];
    const float* bn       = (const float*)d_in[12];
    const float* Wq       = (const float*)d_in[13];
    const float* bq       = (const float*)d_in[14];
    const float* Ws1      = (const float*)d_in[15];
    const float* bs1      = (const float*)d_in[16];
    const float* Ws2      = (const float*)d_in[17];
    const float* bs2      = (const float*)d_in[18];
    float* out = (float*)d_out;

    const int EP = EE + NN;  // edges incl. self loops

    // workspace carve-out (256B aligned)
    char* ws = (char*)d_ws;
    size_t off = 0;
    auto carve = [&](size_t bytes) -> void* {
        void* p = ws + off;
        off = (off + bytes + 255) & ~(size_t)255;
        return p;
    };
    float*    HLIN   = (float*)carve((size_t)NN * HC * 4);
    float*    HACC   = (float*)carve((size_t)NN * HC * 4);
    __bf16*   HB     = (__bf16*)carve((size_t)NN * HC * 2);
    __bf16*   W1t    = (__bf16*)carve((size_t)HC * DD * 2);
    __bf16*   W2t    = (__bf16*)carve((size_t)HC * HC * 2);
    __bf16*   Wnt    = (__bf16*)carve((size_t)HIDD * HC * 2);
    __bf16*   Ws1t   = (__bf16*)carve((size_t)HIDD * HIDD * 2);
    float*    AS     = (float*)carve((size_t)NN * 2 * 4);
    float*    AD     = (float*)carve((size_t)NN * 2 * 4);
    unsigned* SEGMAX = (unsigned*)carve((size_t)NN * 2 * 4);
    float*    SEGSUM = (float*)carve((size_t)NN * 2 * 4);
    float*    ELOG   = (float*)carve((size_t)EP * 2 * 4);
    float*    QEMB   = (float*)carve(128 * 4);
    float*    QVEC   = (float*)carve(128 * 4);

    const int gTile = (NN + 15) / 16;              // 6250
    const int gNode = (NN * HC + 255) / 256;       // 100000
    const int gEdge = (EP + 255) / 256;            // 6641
    const int gAggr = (EP * 32 + 255) / 256;       // 212500
    const int gAlph = (NN * 32 + 255) / 256;       // 12500

    // weight prep
    k_wt<<<(DD * HC + 255) / 256, 256, 0, stream>>>(W1, W1t, DD, HC);
    k_wt<<<(HC * HC + 255) / 256, 256, 0, stream>>>(W2, W2t, HC, HC);
    k_wt<<<(HC * HIDD + 255) / 256, 256, 0, stream>>>(Wn, Wnt, HC, HIDD);
    k_wt<<<(HIDD * HIDD + 255) / 256, 256, 0, stream>>>(Ws1, Ws1t, HIDD, HIDD);  // node half of Ws1
    k_qemb<<<1, 128, 0, stream>>>(q, Wq, bq, QEMB);
    k_qvec<<<1, 128, 0, stream>>>(QEMB, Ws1, bs1, QVEC);

    // ---- GAT layer 1 ----
    k_gemm<float><<<gTile, 256, 0, stream>>>(x, W1t, HLIN, NN, DD, HC);
    k_alpha<<<gAlph, 256, 0, stream>>>(HLIN, a_src1, a_dst1, AS, AD, NN);
    k_init<<<gNode, 256, 0, stream>>>(HACC, SEGMAX, SEGSUM, NN);
    k_elog<<<gEdge, 256, 0, stream>>>(ei, AS, AD, ELOG, SEGMAX, EE, NN);
    k_expsum<<<gEdge, 256, 0, stream>>>(ei, ELOG, SEGMAX, SEGSUM, EE, NN);
    k_agg<<<gAggr, 256, 0, stream>>>(ei, HLIN, ELOG, SEGSUM, HACC, EE, NN);
    k_brelu<<<gNode, 256, 0, stream>>>(HACC, b1, HB, NN);

    // ---- GAT layer 2 ----
    k_gemm<__bf16><<<gTile, 256, 0, stream>>>(HB, W2t, HLIN, NN, HC, HC);
    k_alpha<<<gAlph, 256, 0, stream>>>(HLIN, a_src2, a_dst2, AS, AD, NN);
    k_init<<<gNode, 256, 0, stream>>>(HACC, SEGMAX, SEGSUM, NN);
    k_elog<<<gEdge, 256, 0, stream>>>(ei, AS, AD, ELOG, SEGMAX, EE, NN);
    k_expsum<<<gEdge, 256, 0, stream>>>(ei, ELOG, SEGMAX, SEGSUM, EE, NN);
    k_agg<<<gAggr, 256, 0, stream>>>(ei, HLIN, ELOG, SEGSUM, HACC, EE, NN);
    k_brelu<<<gNode, 256, 0, stream>>>(HACC, b2, HB, NN);

    // ---- fused scorer ----
    const int gFin = (NN + 127) / 128;  // 8 waves x 16 nodes per block
    k_final<<<gFin, 256, 0, stream>>>(HB, Wnt, bn, Ws1t, QVEC, Ws2, bs2, out, NN);

    (void)in_sizes; (void)n_in; (void)out_size; (void)ws_size;
}
